// T5Attention_5952824673130
// MI455X (gfx1250) — compile-verified, matrix operand order checked by reading, not run on hardware
//
#include <hip/hip_runtime.h>
#include <hip/hip_bf16.h>
#include <math.h>

typedef __bf16 bf16;
typedef __attribute__((ext_vector_type(16))) __bf16 v16bf;
typedef __attribute__((ext_vector_type(8)))  __bf16 v8bf;
typedef __attribute__((ext_vector_type(8)))  float  v8f;
typedef unsigned int u32x4 __attribute__((ext_vector_type(4)));
typedef int          i32x4 __attribute__((ext_vector_type(4)));
typedef int          i32x8 __attribute__((ext_vector_type(8)));

#define T5_B   4
#define T5_S   2048
#define T5_H   16
#define T5_D   64
#define T5_DM  1024
#define T5_M   (T5_B * T5_S)      // 8192
#define T5_BH  (T5_B * T5_H)      // 64
#define BIAS_W (2 * T5_S - 1)     // 4095

#if __has_builtin(__builtin_amdgcn_tensor_load_to_lds)
#define HAVE_TDM 1
#else
#define HAVE_TDM 0
#endif

// ---------------------------------------------------------------- WMMA helpers

__device__ __forceinline__ v8f wmma_bf16f32(v16bf a, v16bf b, v8f c) {
  // D = A(16x32 bf16) * B(32x16 bf16) + C(16x16 f32)
  return __builtin_amdgcn_wmma_f32_16x16x32_bf16(
      /*neg_a=*/false, a, /*neg_b=*/false, b,
      /*c_mod=*/(short)0, c, /*reuse_a=*/false, /*reuse_b=*/false);
}

// Load a 16x32 bf16 fragment (A layout; B layout identical when the source is
// stored "n-major") from row-major memory (global or LDS).
// Lane L (L<16): row=L, halves = [k0..k0+7] and [k0+16..k0+23]
// Lane L+16:     row=L, halves = [k0+8..k0+15] and [k0+24..k0+31]
__device__ __forceinline__ v16bf ldfrag(const bf16* base, int stride,
                                        int row0, int k0, int lane) {
  int r    = lane & 15;
  int koff = (lane >> 4) << 3;
  const bf16* p = base + (size_t)(row0 + r) * stride + (k0 + koff);
  union { v16bf v; v8bf h[2]; } u;
  u.h[0] = *(const v8bf*)(p);
  u.h[1] = *(const v8bf*)(p + 16);
  return u.v;
}

// ---------------------------------------------------------------- TDM helpers

#if HAVE_TDM
// Issue TENSOR_LOAD_TO_LDS of a 2D tile (2-byte elements).
//   tile = tile_d1 rows x tile_d0 elements, row stride = stride0 elements,
//   packed contiguously into LDS at lds_addr.
// D# layout per CDNA5 ISA ch.8 (groups 2/3 zero => 2D tensor).
__device__ __forceinline__ void tdm_load_2d(unsigned lds_addr, const void* gptr,
                                            unsigned tensor_d0, unsigned tensor_d1,
                                            unsigned tile_d0, unsigned tile_d1,
                                            unsigned long long stride0) {
  unsigned long long ga = (unsigned long long)(uintptr_t)gptr;
  u32x4 g0;
  g0[0] = 1u;                                              // count=1 (valid, user)
  g0[1] = lds_addr;                                        // lds_addr[31:0]
  g0[2] = (unsigned)(ga & 0xFFFFFFFFu);                    // global_addr lo
  g0[3] = (unsigned)((ga >> 32) & 0x01FFFFFFu) | (2u << 30); // addr hi | type=2
  i32x8 g1;
  g1[0] = (int)(1u << 16);                                 // data_size=1 (2 bytes)
  g1[1] = (int)((tensor_d0 & 0xFFFFu) << 16);              // tensor_dim0[15:0]
  g1[2] = (int)((tensor_d0 >> 16) | ((tensor_d1 & 0xFFFFu) << 16));
  g1[3] = (int)((tensor_d1 >> 16) | ((tile_d0 & 0xFFFFu) << 16));
  g1[4] = (int)(tile_d1 & 0xFFFFu);                        // tile_dim1, tile_dim2=0
  g1[5] = (int)(unsigned)(stride0 & 0xFFFFFFFFu);          // dim0_stride lo
  g1[6] = (int)(unsigned)((stride0 >> 32) & 0xFFFFu);      // dim0_stride hi
  g1[7] = 0;
  i32x4 gz = {0, 0, 0, 0};
#if defined(__clang_major__) && (__clang_major__ >= 23)
  i32x8 gz8 = {0, 0, 0, 0, 0, 0, 0, 0};
  __builtin_amdgcn_tensor_load_to_lds(g0, g1, gz, gz, gz8, 0);
#else
  __builtin_amdgcn_tensor_load_to_lds(g0, g1, gz, gz, 0);
#endif
}

__device__ __forceinline__ void tdm_wait_le2() {
#if __has_builtin(__builtin_amdgcn_s_wait_tensorcnt)
  __builtin_amdgcn_s_wait_tensorcnt(2);
#else
  asm volatile("s_wait_tensorcnt 0x2" ::: "memory");
#endif
}
__device__ __forceinline__ void tdm_wait_0() {
#if __has_builtin(__builtin_amdgcn_s_wait_tensorcnt)
  __builtin_amdgcn_s_wait_tensorcnt(0);
#else
  asm volatile("s_wait_tensorcnt 0x0" ::: "memory");
#endif
}
#endif  // HAVE_TDM

// ---------------------------------------------------------------- conversions

__global__ __launch_bounds__(256) void t5_cvt_bf16(const float* __restrict__ x,
                                                   bf16* __restrict__ y, int n) {
  int i = blockIdx.x * 256 + threadIdx.x;
  if (i < n) y[i] = (bf16)x[i];
}

// Wt[n][k] = (bf16) W[k][n]
__global__ __launch_bounds__(256) void t5_transpose_bf16(const float* __restrict__ W,
                                                         bf16* __restrict__ Wt,
                                                         int K, int N) {
  int i = blockIdx.x * 256 + threadIdx.x;
  if (i >= K * N) return;
  int k = i / N, n = i % N;
  Wt[(size_t)n * K + k] = (bf16)W[(size_t)k * N + n];
}

// bias_tab[h][delta + S-1] = rel_bias[bucket(delta)][h], delta = key - query
__global__ __launch_bounds__(256) void t5_bias_table(const float* __restrict__ rel_bias,
                                                     float* __restrict__ tab) {
  int i = blockIdx.x * 256 + threadIdx.x;
  if (i >= T5_H * BIAS_W) return;
  int h = i / BIAS_W;
  int delta = (i % BIAS_W) - (T5_S - 1);
  int rb = (delta > 0) ? 16 : 0;
  int rp = (delta < 0) ? -delta : delta;
  int bucket;
  if (rp < 8) {
    bucket = rb + rp;
  } else {
    float l = __logf((float)rp * 0.125f) * (8.0f / __logf(16.0f));
    int li = 8 + (int)l;
    if (li > 15) li = 15;
    bucket = rb + li;
  }
  tab[i] = rel_bias[bucket * T5_H + h];
}

// ---------------------------------------------------------------- GEMM (bf16 WMMA)
// mode 0: bf16 Q/K layout [bh][s][64]; mode 1: bf16 Vt layout [bh][64][s];
// mode 2: f32 row-major [M][N]
__global__ __launch_bounds__(256) void t5_gemm_bf16(const bf16* __restrict__ A,
                                                    const bf16* __restrict__ Bt,
                                                    void* __restrict__ Cp,
                                                    int M, int N, int K, int mode) {
  int lane = threadIdx.x & 31;
  int w    = threadIdx.x >> 5;
  int wm = w >> 1, wn = w & 1;
  int m0 = blockIdx.y * 128 + wm * 32;
  int n0 = blockIdx.x * 64  + wn * 32;

  v8f acc[2][2] = {};
  for (int k0 = 0; k0 < K; k0 += 32) {
    v16bf a0 = ldfrag(A,  K, m0,      k0, lane);
    v16bf a1 = ldfrag(A,  K, m0 + 16, k0, lane);
    v16bf b0 = ldfrag(Bt, K, n0,      k0, lane);
    v16bf b1 = ldfrag(Bt, K, n0 + 16, k0, lane);
    acc[0][0] = wmma_bf16f32(a0, b0, acc[0][0]);
    acc[0][1] = wmma_bf16f32(a0, b1, acc[0][1]);
    acc[1][0] = wmma_bf16f32(a1, b0, acc[1][0]);
    acc[1][1] = wmma_bf16f32(a1, b1, acc[1][1]);
  }

  int mbase = (lane < 16) ? 0 : 8;
  int colb  = lane & 15;
#pragma unroll
  for (int i = 0; i < 2; ++i)
#pragma unroll
    for (int j = 0; j < 2; ++j)
#pragma unroll
      for (int r = 0; r < 8; ++r) {
        int m = m0 + i * 16 + mbase + r;
        int n = n0 + j * 16 + colb;
        float val = acc[i][j][r];
        if (mode == 2) {
          ((float*)Cp)[(size_t)m * N + n] = val;
        } else {
          int b = m >> 11, s = m & (T5_S - 1);
          int h = n >> 6,  d = n & (T5_D - 1);
          size_t idx = (mode == 0)
              ? (((size_t)(b * T5_H + h) * T5_S + s) * T5_D + d)
              : (((size_t)(b * T5_H + h) * T5_D + d) * T5_S + s);
          ((bf16*)Cp)[idx] = (bf16)val;
        }
      }
}

// ---------------------------------------------------------------- fused attention
// grid = (S/128, B*H); 8 waves/block, one 16-row Q tile per wave.
// K/V tiles double-buffered in LDS via the Tensor Data Mover (wave 0 issues,
// TENSORcnt + workgroup barriers synchronize); flash softmax over 32-key tiles.
__global__ __launch_bounds__(256) void t5_flash_attn(
    const bf16* __restrict__ q,         // [bh][s][64]
    const bf16* __restrict__ k,         // [bh][s][64]
    const bf16* __restrict__ vt,        // [bh][64][s]
    const float* __restrict__ bias_tab, // [h][4095]
    bf16* __restrict__ ctx) {           // [b*s][1024], col = h*64+d
  __shared__ __align__(16) bf16 k_lds[2][32][T5_D];   // 8 KB
  __shared__ __align__(16) bf16 v_lds[2][T5_D][32];   // 8 KB
  __shared__ __align__(16) bf16 p_lds[8][16 * 32];    // 8 KB

  int lane = threadIdx.x & 31;
  int w    = threadIdx.x >> 5;
  int bh = blockIdx.y;
  int b  = bh >> 4, h = bh & 15;
  int q0 = blockIdx.x * 128 + w * 16;

  const bf16*  qbh  = q  + (size_t)bh * T5_S * T5_D;
  const bf16*  kbh  = k  + (size_t)bh * T5_S * T5_D;
  const bf16*  vbh  = vt + (size_t)bh * T5_D * T5_S;
  const float* btab = bias_tab + (size_t)h * BIAS_W + (T5_S - 1);

  // Q fragments stay in registers for the whole kernel.
  v16bf qa0 = ldfrag(qbh, T5_D, q0, 0,  lane);
  v16bf qa1 = ldfrag(qbh, T5_D, q0, 32, lane);

  float mrun[8], lrun[8];
  v8f acc[4] = {};
#pragma unroll
  for (int r = 0; r < 8; ++r) { mrun[r] = -1e30f; lrun[r] = 0.0f; }

  int mbase = (lane < 16) ? 0 : 8;
  int colb  = lane & 15;

  const int NT = T5_S / 32;  // 64 key tiles

#if HAVE_TDM
  // Stage a K tile (32 rows x 64) and a V tile (64 rows x 32) for key tile tt.
  auto stage = [&](int tt) {
    int buf = tt & 1;
    tdm_load_2d((unsigned)(uintptr_t)&k_lds[buf][0][0], kbh + (size_t)tt * 32 * T5_D,
                /*tensor_d0=*/T5_D, /*tensor_d1=*/T5_S,
                /*tile_d0=*/T5_D, /*tile_d1=*/32, /*stride0=*/T5_D);
    tdm_load_2d((unsigned)(uintptr_t)&v_lds[buf][0][0], vbh + (size_t)tt * 32,
                /*tensor_d0=*/T5_S, /*tensor_d1=*/T5_D,
                /*tile_d0=*/32, /*tile_d1=*/T5_D, /*stride0=*/T5_S);
  };
  if (w == 0) { stage(0); stage(1); }
#else
  // Cooperative fallback: all 256 threads copy the tiles.
  auto stage = [&](int tt) {
    int buf = tt & 1;
    int tid = threadIdx.x;
    {  // K tile is contiguous (row length == stride): flat 4KB copy
      v8bf* dst = (v8bf*)&k_lds[buf][0][0];
      const v8bf* src = (const v8bf*)(kbh + (size_t)tt * 32 * T5_D);
      dst[tid] = src[tid];
    }
    {  // V tile: 64 rows of 32 elements, row stride T5_S
      int r = tid >> 2, c = (tid & 3) * 8;
      *(v8bf*)&v_lds[buf][r][c] = *(const v8bf*)(vbh + (size_t)r * T5_S + tt * 32 + c);
    }
  };
  stage(0); stage(1);
#endif

  for (int t = 0; t < NT; ++t) {
    int j   = t * 32;
    int buf = t & 1;
#if HAVE_TDM
    if (w == 0) {
      if (t + 1 < NT) tdm_wait_le2();  // oldest 2 (this tile's K,V) complete
      else            tdm_wait_0();
    }
#endif
    __syncthreads();
    const bf16* kt = &k_lds[buf][0][0];
    const bf16* vl = &v_lds[buf][0][0];

    // ---- scores: S[16x32] = Q[16x64] @ K_tile^T, 4 WMMAs (B frags from LDS)
    v8f s[2];
#pragma unroll
    for (int nt = 0; nt < 2; ++nt) {
      v16bf kb0 = ldfrag(kt, T5_D, nt * 16, 0,  lane);
      v16bf kb1 = ldfrag(kt, T5_D, nt * 16, 32, lane);
      v8f c = {};
      c = wmma_bf16f32(qa0, kb0, c);
      c = wmma_bf16f32(qa1, kb1, c);
      s[nt] = c;
    }
    // ---- relative-position bias
#pragma unroll
    for (int nt = 0; nt < 2; ++nt)
#pragma unroll
      for (int r = 0; r < 8; ++r) {
        int m   = mbase + r;
        int key = j + nt * 16 + colb;
        s[nt][r] += btab[key - (q0 + m)];
      }
    // ---- online softmax (row stats live in each 16-lane half)
    float corr[8];
#pragma unroll
    for (int r = 0; r < 8; ++r) {
      float mx = fmaxf(s[0][r], s[1][r]);
#pragma unroll
      for (int off = 1; off < 16; off <<= 1)
        mx = fmaxf(mx, __shfl_xor(mx, off, 32));
      float mnew = fmaxf(mrun[r], mx);
      corr[r] = __expf(mrun[r] - mnew);
      mrun[r] = mnew;
      float p0 = __expf(s[0][r] - mnew);
      float p1 = __expf(s[1][r] - mnew);
      s[0][r] = p0; s[1][r] = p1;
      float rs = p0 + p1;
#pragma unroll
      for (int off = 1; off < 16; off <<= 1)
        rs += __shfl_xor(rs, off, 32);
      lrun[r] = lrun[r] * corr[r] + rs;
    }
#pragma unroll
    for (int dt = 0; dt < 4; ++dt)
#pragma unroll
      for (int r = 0; r < 8; ++r)
        acc[dt][r] *= corr[r];

    // ---- re-fragment P through LDS (D layout -> A layout)
    bf16* pl = &p_lds[w][0];
#pragma unroll
    for (int nt = 0; nt < 2; ++nt)
#pragma unroll
      for (int r = 0; r < 8; ++r)
        pl[(mbase + r) * 32 + nt * 16 + colb] = (bf16)s[nt][r];
    asm volatile("s_wait_dscnt 0x0" ::: "memory");  // cross-lane LDS RAW in-wave

    {
      int koff = (lane >> 4) << 3;
      const bf16* pp = &p_lds[w][(lane & 15) * 32 + koff];
      union { v16bf v; v8bf hh[2]; } up;
      up.hh[0] = *(const v8bf*)(pp);
      up.hh[1] = *(const v8bf*)(pp + 16);
      // ---- ctx += P[16x32] @ V_tile, 4 WMMAs (B frags from staged Vt in LDS)
#pragma unroll
      for (int dt = 0; dt < 4; ++dt) {
        v16bf vb = ldfrag(vl, 32, dt * 16, 0, lane);
        acc[dt] = wmma_bf16f32(up.v, vb, acc[dt]);
      }
    }
    __syncthreads();  // everyone done reading buf before it is overwritten
#if HAVE_TDM
    if (w == 0 && t + 2 < NT) stage(t + 2);
#else
    if (t + 2 < NT) stage(t + 2);
#endif
  }

  // ---- epilogue: normalize, store bf16 ctx for the output projection
#pragma unroll
  for (int dt = 0; dt < 4; ++dt)
#pragma unroll
    for (int r = 0; r < 8; ++r) {
      int m = mbase + r;
      int d = dt * 16 + colb;
      float val = acc[dt][r] / lrun[r];
      size_t row = (size_t)b * T5_S + (q0 + m);
      ctx[row * (size_t)T5_DM + h * T5_D + d] = (bf16)val;
    }
}

// ---------------------------------------------------------------- launcher

extern "C" void kernel_launch(void* const* d_in, const int* in_sizes, int n_in,
                              void* d_out, int out_size, void* d_ws, size_t ws_size,
                              hipStream_t stream) {
  (void)in_sizes; (void)n_in; (void)out_size; (void)ws_size;
  const float* hs  = (const float*)d_in[0];
  const float* wq  = (const float*)d_in[1];
  const float* wk  = (const float*)d_in[2];
  const float* wv  = (const float*)d_in[3];
  const float* wo  = (const float*)d_in[4];
  const float* rel = (const float*)d_in[5];

  char* ws = (char*)d_ws;
  const size_t SZ_HID = (size_t)T5_M * T5_DM * sizeof(bf16);
  const size_t SZ_W   = (size_t)T5_DM * T5_DM * sizeof(bf16);
  const size_t SZ_QKV = (size_t)T5_BH * T5_S * T5_D * sizeof(bf16);

  bf16* hb   = (bf16*)(ws);
  bf16* wqt  = (bf16*)(ws + SZ_HID);
  bf16* wkt  = (bf16*)(ws + SZ_HID + SZ_W);
  bf16* wvt  = (bf16*)(ws + SZ_HID + 2 * SZ_W);
  bf16* wot  = (bf16*)(ws + SZ_HID + 3 * SZ_W);
  bf16* qb   = (bf16*)(ws + SZ_HID + 4 * SZ_W);
  bf16* kb   = (bf16*)(ws + SZ_HID + 4 * SZ_W + SZ_QKV);
  bf16* vtb  = (bf16*)(ws + SZ_HID + 4 * SZ_W + 2 * SZ_QKV);
  bf16* ctxb = (bf16*)(ws + SZ_HID + 4 * SZ_W + 3 * SZ_QKV);
  float* btab = (float*)(ws + SZ_HID + 4 * SZ_W + 3 * SZ_QKV + SZ_HID);

  {
    int n = T5_M * T5_DM;
    t5_cvt_bf16<<<(n + 255) / 256, 256, 0, stream>>>(hs, hb, n);
    int nw = T5_DM * T5_DM;
    t5_transpose_bf16<<<(nw + 255) / 256, 256, 0, stream>>>(wq, wqt, T5_DM, T5_DM);
    t5_transpose_bf16<<<(nw + 255) / 256, 256, 0, stream>>>(wk, wkt, T5_DM, T5_DM);
    t5_transpose_bf16<<<(nw + 255) / 256, 256, 0, stream>>>(wv, wvt, T5_DM, T5_DM);
    t5_transpose_bf16<<<(nw + 255) / 256, 256, 0, stream>>>(wo, wot, T5_DM, T5_DM);
    int nb = T5_H * BIAS_W;
    t5_bias_table<<<(nb + 255) / 256, 256, 0, stream>>>(rel, btab);
  }

  dim3 ggrid(T5_DM / 64, T5_M / 128);
  t5_gemm_bf16<<<ggrid, 256, 0, stream>>>(hb, wqt, qb,  T5_M, T5_DM, T5_DM, 0);
  t5_gemm_bf16<<<ggrid, 256, 0, stream>>>(hb, wkt, kb,  T5_M, T5_DM, T5_DM, 0);
  t5_gemm_bf16<<<ggrid, 256, 0, stream>>>(hb, wvt, vtb, T5_M, T5_DM, T5_DM, 1);

  dim3 agrid(T5_S / 128, T5_BH);
  t5_flash_attn<<<agrid, 256, 0, stream>>>(qb, kb, vtb, btab, ctxb);

  t5_gemm_bf16<<<ggrid, 256, 0, stream>>>(ctxb, wot, (float*)d_out,
                                          T5_M, T5_DM, T5_DM, 2);
}
